// TransformerBlock_73641509257695
// MI455X (gfx1250) — matrix operand order repaired and checked by reading, not compiled, not run on hardware
//
#include <hip/hip_runtime.h>
#include <hip/hip_bf16.h>
#include <math.h>

typedef __attribute__((ext_vector_type(8)))  _Float16 v8h;
typedef __attribute__((ext_vector_type(16))) _Float16 v16h;
typedef __attribute__((ext_vector_type(8)))  float    v8f;

#define DEVFN static __device__ __forceinline__

// ---------------------------------------------------------------------------
// CDNA5 WMMA helpers (wave32, V_WMMA_F32_16X16X32_F16)
// ---------------------------------------------------------------------------
DEVFN v8f wmma_f16(v16h a, v16h b, v8f c) {
  // 8 args: (neg_a, A, neg_b, B, c_mod, C, reuse_a, reuse_b)
  return __builtin_amdgcn_wmma_f32_16x16x32_f16(false, a, false, b, (short)0, c,
                                                false, false);
}

// A fragment: 16x32 f16 tile from row-major source (leading dim lda, elems).
// Layout: lane L<16 holds row M=L, K = 0..15;
// lane L>=16 holds row M=L-16, K = 16..31.  -> two contiguous v8h.
DEVFN v16h load_a_frag(const _Float16* A, int lda, int k) {
  const int lane = threadIdx.x & 31;
  const int half = lane >> 4;
  const int m    = lane & 15;
  const _Float16* row = A + m * lda + k + 16 * half;
  union { v16h v; v8h h[2]; } u;
  u.h[0] = *(const v8h*)(row);
  u.h[1] = *(const v8h*)(row + 8);
  return u.v;
}

// B fragment: 32x16 f16 tile taken from an N-major matrix Bt[n*ldk + k] (i.e.
// Bt is B transposed / column-major B).  Lane L<16: column N=L, K=k..k+15;
// lane L>=16: column N=L-16, K=k+16..k+31.  -> one contiguous 32B load.
DEVFN v16h load_bT_frag(const _Float16* Bt, int ldk, int k) {
  const int lane = threadIdx.x & 31;
  const int half = lane >> 4;
  const int n    = lane & 15;
  return *(const v16h*)(Bt + n * ldk + k + 16 * half);
}

// 2x2 register-blocked tile: 32x32 output per wave, A/B fragments reused 2x.
// A: rows m0..m0+31 (row-major, ld=K);  Bt: rows n0..n0+31 (N-major, ld=K).
DEVFN void gemm_tile_2x2(const _Float16* A, const _Float16* Bt, int K,
                         v8f& c00, v8f& c01, v8f& c10, v8f& c11) {
  for (int k = 0; k < K; k += 32) {
    v16h a0 = load_a_frag(A, K, k);
    v16h a1 = load_a_frag(A + 16 * K, K, k);
    v16h b0 = load_bT_frag(Bt, K, k);
    v16h b1 = load_bT_frag(Bt + 16 * K, K, k);
    c00 = wmma_f16(a0, b0, c00);
    c01 = wmma_f16(a0, b1, c01);
    c10 = wmma_f16(a1, b0, c10);
    c11 = wmma_f16(a1, b1, c11);
  }
}

// C/D layout: VGPR r, lane: n = lane&15, row m = r + 8*(lane>>4).

// ---------------------------------------------------------------------------
// Elementwise prep kernels
// ---------------------------------------------------------------------------

// (B,C,H,W) f32 -> (B*1024, 192) f16 patches, j = (p1*8+p2)*3 + c
__global__ void patchify_kernel(const float* __restrict__ img,
                                _Float16* __restrict__ out) {
  int idx = blockIdx.x * 256 + threadIdx.x;
  if (idx >= 8 * 1024 * 192) return;
  int j = idx % 192;
  int p = (idx / 192) & 1023;
  int b = idx / (192 * 1024);
  int p1 = j / 24, rem = j % 24, p2 = rem / 3, c = rem % 3;
  int hp = p >> 5, wp = p & 31;
  out[idx] = (_Float16)img[((b * 3 + c) * 256 + hp * 8 + p1) * 256 + wp * 8 + p2];
}

// w (KxN f32, row-major) -> wt (NxK f16)  : fragment-friendly weight staging
__global__ void cast_transpose_kernel(const float* __restrict__ w,
                                      _Float16* __restrict__ wt, int K, int N) {
  int idx = blockIdx.x * 256 + threadIdx.x;
  if (idx >= K * N) return;
  int k = idx / N, n = idx % N;
  wt[n * K + k] = (_Float16)w[idx];
}

// mask (B,1,256,256) -> per-(b,patch) additive bias {0,-100}
__global__ void maskbias_kernel(const float* __restrict__ mask,
                                float* __restrict__ biasb) {
  int idx = blockIdx.x * 256 + threadIdx.x;
  if (idx >= 8192) return;
  int b = idx >> 10, p = idx & 1023;
  int hp = p >> 5, wp = p & 31;
  const float* m = mask + (b * 256 + hp * 8) * 256 + wp * 8;
  float s = 0.f;
  for (int i = 0; i < 8; ++i)
    for (int j = 0; j < 8; ++j) s += m[i * 256 + j];
  biasb[idx] = (s > 0.f) ? 0.f : -100.f;
}

// ---------------------------------------------------------------------------
// GEMM kernels: one wave per 32x32 output tile, 4 waves / 128-thread block
// ---------------------------------------------------------------------------

DEVFN void embed_store(v8f accT, v8f accB, int col, int m0, int half,
                       const float* bias, _Float16* X) {
  float div = __expf((float)(col & ~1) * (-9.210340371976184f / 256.0f));
  float bv = bias[col];
#pragma unroll
  for (int r = 0; r < 8; ++r) {
#pragma unroll
    for (int i = 0; i < 2; ++i) {
      int m = m0 + 16 * i + r + 8 * half;
      float pos = (float)(m & 1023);
      float pe = (col & 1) ? __cosf(pos * div) : __sinf(pos * div);
      float a = (i ? accB[r] : accT[r]);
      X[m * 256 + col] = (_Float16)(a + bv + pe);
    }
  }
}

// patches (8192x192) @ w_patch (192x256) + b + sinusoidal PE -> x f16
__global__ __launch_bounds__(128)
void embed_kernel(const _Float16* __restrict__ A, const _Float16* __restrict__ Bt,
                  const float* __restrict__ bias, _Float16* __restrict__ X) {
  int wid = blockIdx.x * 4 + (threadIdx.x >> 5);
  int m0 = (wid >> 3) * 32, n0 = (wid & 7) * 32;
  v8f c00 = {}, c01 = {}, c10 = {}, c11 = {};
  gemm_tile_2x2(A + m0 * 192, Bt + n0 * 192, 192, c00, c01, c10, c11);
  int lane = threadIdx.x & 31, half = lane >> 4, n = lane & 15;
  embed_store(c00, c10, n0 + n,      m0, half, bias, X);
  embed_store(c01, c11, n0 + 16 + n, m0, half, bias, X);
}

DEVFN void kv_store(v8f accT, v8f accB, int col, int m0, int half,
                    const float* bias, _Float16* Kb, _Float16* Vt) {
  int jj = col & 255, head = jj >> 5, hd = jj & 31;
  bool isV = col >= 256;
  float bv = bias[col];
#pragma unroll
  for (int r = 0; r < 8; ++r) {
#pragma unroll
    for (int i = 0; i < 2; ++i) {
      int m = m0 + 16 * i + r + 8 * half;
      int b = m >> 10, tok = m & 1023;
      float v = (i ? accB[r] : accT[r]) + bv;
      if (isV)
        Vt[((b * 8 + head) * 32 + hd) * 1024 + tok] = (_Float16)v;  // V^T
      else
        Kb[((b * 8 + head) * 1024 + tok) * 32 + hd] = (_Float16)v;
    }
  }
}

// x0 (8192x256) @ w_kv (256x512): cols<256 -> K (b,h,tok,32); cols>=256 -> Vt (b,h,hd,tok)
__global__ __launch_bounds__(128)
void kv_kernel(const _Float16* __restrict__ A, const _Float16* __restrict__ Bt,
               const float* __restrict__ bias, _Float16* __restrict__ Kb,
               _Float16* __restrict__ Vt) {
  int wid = blockIdx.x * 4 + (threadIdx.x >> 5);
  int m0 = (wid >> 4) * 32, n0 = (wid & 15) * 32;
  v8f c00 = {}, c01 = {}, c10 = {}, c11 = {};
  gemm_tile_2x2(A + m0 * 256, Bt + n0 * 256, 256, c00, c01, c10, c11);
  int lane = threadIdx.x & 31, half = lane >> 4, n = lane & 15;
  kv_store(c00, c10, n0 + n,      m0, half, bias, Kb, Vt);
  kv_store(c01, c11, n0 + 16 + n, m0, half, bias, Kb, Vt);
}

DEVFN void q_store(v8f accT, v8f accB, int col, int m0, int half,
                   const float* bias, _Float16* Qb) {
  int head = col >> 5, hd = col & 31;
  float bv = bias[col];
#pragma unroll
  for (int r = 0; r < 8; ++r) {
#pragma unroll
    for (int i = 0; i < 2; ++i) {
      int m = m0 + 16 * i + r + 8 * half;
      int b = m >> 10, tok = m & 1023;
      float v = (i ? accB[r] : accT[r]) + bv;
      Qb[((b * 8 + head) * 1024 + tok) * 32 + hd] = (_Float16)v;
    }
  }
}

// x1 (8192x256) @ w_qv[:, :256] -> Q (b,h,tok,32)
__global__ __launch_bounds__(128)
void q_kernel(const _Float16* __restrict__ A, const _Float16* __restrict__ Bt,
              const float* __restrict__ bias, _Float16* __restrict__ Qb) {
  int wid = blockIdx.x * 4 + (threadIdx.x >> 5);
  int m0 = (wid >> 3) * 32, n0 = (wid & 7) * 32;
  v8f c00 = {}, c01 = {}, c10 = {}, c11 = {};
  gemm_tile_2x2(A + m0 * 256, Bt + n0 * 256, 256, c00, c01, c10, c11);
  int lane = threadIdx.x & 31, half = lane >> 4, n = lane & 15;
  q_store(c00, c10, n0 + n,      m0, half, bias, Qb);
  q_store(c01, c11, n0 + 16 + n, m0, half, bias, Qb);
}

// ---------------------------------------------------------------------------
// Flash attention: one wave = one (b,h,32-row q block); K/V fragments are
// loaded once per 32-key step and reused by both q tiles (halves K/V traffic).
// ---------------------------------------------------------------------------

DEVFN void flash_update(v8f s0, v8f s1, float b0, float b1,
                        float* mrow, float* lrow, v8f& acc0, v8f& acc1,
                        _Float16* lp, int half, int n) {
  const float SCL = 0.17677669529663687f;  // 32^-0.5
#pragma unroll
  for (int r = 0; r < 8; ++r) {
    float t0 = s0[r] * SCL + b0;
    float t1 = s1[r] * SCL + b1;
    float t = fmaxf(t0, t1);
#pragma unroll
    for (int mm = 1; mm < 16; mm <<= 1) t = fmaxf(t, __shfl_xor(t, mm, 32));
    float mn = fmaxf(mrow[r], t);
    float corr = __expf(mrow[r] - mn);
    mrow[r] = mn;
    float p0 = __expf(t0 - mn), p1 = __expf(t1 - mn);
    float ps = p0 + p1;
#pragma unroll
    for (int mm = 1; mm < 16; mm <<= 1) ps += __shfl_xor(ps, mm, 32);
    lrow[r] = lrow[r] * corr + ps;
    acc0[r] *= corr; acc1[r] *= corr;
    int mr = r + 8 * half;                 // C-layout row of this lane
    lp[mr * 32 + n]      = (_Float16)p0;
    lp[mr * 32 + 16 + n] = (_Float16)p1;
  }
}

__global__ __launch_bounds__(128)
void attn_kernel(const _Float16* __restrict__ Qb, const _Float16* __restrict__ Kb,
                 const _Float16* __restrict__ Vt, const float* __restrict__ biasb,
                 _Float16* __restrict__ O) {
  __shared__ __align__(32) _Float16 lds[4 * 2 * 512];   // 2 p-tiles per wave
  int wid = blockIdx.x * 4 + (threadIdx.x >> 5);        // 0..2047
  int qt = wid & 31, bh = wid >> 5;                     // bh = b*8 + head
  int b = bh >> 3;
  const _Float16* Q  = Qb + (bh * 1024 + qt * 32) * 32;
  const _Float16* Kp = Kb + bh * 1024 * 32;
  const _Float16* Vp = Vt + bh * 32 * 1024;
  const float* brow  = biasb + b * 1024;
  int lane = threadIdx.x & 31, half = lane >> 4, n = lane & 15;
  _Float16* lp0 = lds + (threadIdx.x >> 5) * 1024;
  _Float16* lp1 = lp0 + 512;

  v16h qf0 = load_a_frag(Q, 32, 0);          // K=32 == head_dim: load once
  v16h qf1 = load_a_frag(Q + 16 * 32, 32, 0);
  v8f acc00 = {}, acc01 = {}, acc10 = {}, acc11 = {};
  float mrow0[8], lrow0[8], mrow1[8], lrow1[8];
#pragma unroll
  for (int r = 0; r < 8; ++r) {
    mrow0[r] = -1e30f; lrow0[r] = 0.f;
    mrow1[r] = -1e30f; lrow1[r] = 0.f;
  }

  for (int k0 = 0; k0 < 1024; k0 += 32) {
    // k^T fragments: K rows are contiguous (hd-major), 32B per lane
    v16h kb0 = *(const v16h*)(Kp + (k0 + n) * 32 + 16 * half);
    v16h kb1 = *(const v16h*)(Kp + (k0 + 16 + n) * 32 + 16 * half);
    v8f z = {};
    v8f s00 = wmma_f16(qf0, kb0, z);
    v8f s01 = wmma_f16(qf0, kb1, z);
    v8f s10 = wmma_f16(qf1, kb0, z);
    v8f s11 = wmma_f16(qf1, kb1, z);
    float b0 = brow[k0 + n], b1 = brow[k0 + 16 + n];
    flash_update(s00, s01, b0, b1, mrow0, lrow0, acc00, acc01, lp0, half, n);
    flash_update(s10, s11, b0, b1, mrow1, lrow1, acc10, acc11, lp1, half, n);
    asm volatile("s_wait_dscnt 0x0" ::: "memory");   // LDS stores visible
    v16h pf0 = load_a_frag(lp0, 32, 0);       // probs as A-fragments (16x32)
    v16h pf1 = load_a_frag(lp1, 32, 0);
    // V fragments: Vt is hd-major so K (token) dim is contiguous
    v16h vb0 = *(const v16h*)(Vp + n * 1024 + k0 + 16 * half);
    v16h vb1 = *(const v16h*)(Vp + (16 + n) * 1024 + k0 + 16 * half);
    acc00 = wmma_f16(pf0, vb0, acc00);
    acc01 = wmma_f16(pf0, vb1, acc01);
    acc10 = wmma_f16(pf1, vb0, acc10);
    acc11 = wmma_f16(pf1, vb1, acc11);
  }
#pragma unroll
  for (int r = 0; r < 8; ++r) {
    float inv0 = 1.f / lrow0[r];
    float inv1 = 1.f / lrow1[r];
    int tok0 = qt * 32 + r + 8 * half;
    int tok1 = tok0 + 16;
    _Float16* d0 = O + (b * 1024 + tok0) * 256 + (bh & 7) * 32;
    _Float16* d1 = O + (b * 1024 + tok1) * 256 + (bh & 7) * 32;
    d0[n]      = (_Float16)(acc00[r] * inv0);
    d0[16 + n] = (_Float16)(acc01[r] * inv0);
    d1[n]      = (_Float16)(acc10[r] * inv1);
    d1[16 + n] = (_Float16)(acc11[r] * inv1);
  }
}

// ---------------------------------------------------------------------------
// MLP
// ---------------------------------------------------------------------------

DEVFN void fc1_store(v8f accT, v8f accB, int col, int m0, int half,
                     const float* bias, _Float16* H) {
  float bv = bias[col];
#pragma unroll
  for (int r = 0; r < 8; ++r) {
#pragma unroll
    for (int i = 0; i < 2; ++i) {
      float x = (i ? accB[r] : accT[r]) + bv;
      float g = 0.5f * x * (1.f + erff(x * 0.7071067811865475f));  // exact GELU
      H[(m0 + 16 * i + r + 8 * half) * 1024 + col] = (_Float16)g;
    }
  }
}

__global__ __launch_bounds__(128)
void fc1_kernel(const _Float16* __restrict__ A, const _Float16* __restrict__ Bt,
                const float* __restrict__ bias, _Float16* __restrict__ H) {
  int wid = blockIdx.x * 4 + (threadIdx.x >> 5);
  int m0 = (wid >> 5) * 32, n0 = (wid & 31) * 32;
  v8f c00 = {}, c01 = {}, c10 = {}, c11 = {};
  gemm_tile_2x2(A + m0 * 256, Bt + n0 * 256, 256, c00, c01, c10, c11);
  int lane = threadIdx.x & 31, half = lane >> 4, n = lane & 15;
  fc1_store(c00, c10, n0 + n,      m0, half, bias, H);
  fc1_store(c01, c11, n0 + 16 + n, m0, half, bias, H);
}

DEVFN void fc2_store(v8f accT, v8f accB, int col, int m0, int half,
                     const float* bias, float* out) {
  float bv = bias[col];
  int p1 = col / 24, rem = col % 24, p2 = rem / 3, c = rem % 3;
#pragma unroll
  for (int r = 0; r < 8; ++r) {
#pragma unroll
    for (int i = 0; i < 2; ++i) {
      int m = m0 + 16 * i + r + 8 * half;
      int b = m >> 10, p = m & 1023, hp = p >> 5, wp = p & 31;
      out[((b * 3 + c) * 256 + hp * 8 + p1) * 256 + wp * 8 + p2] =
          (i ? accB[r] : accT[r]) + bv;
    }
  }
}

// h (8192x1024) @ w_fc2 (1024x192) + b -> unpatchify scatter into (B,3,256,256)
__global__ __launch_bounds__(128)
void fc2_kernel(const _Float16* __restrict__ A, const _Float16* __restrict__ Bt,
                const float* __restrict__ bias, float* __restrict__ out) {
  int wid = blockIdx.x * 4 + (threadIdx.x >> 5);
  int m0 = (wid / 6) * 32, n0 = (wid % 6) * 32;
  v8f c00 = {}, c01 = {}, c10 = {}, c11 = {};
  gemm_tile_2x2(A + m0 * 1024, Bt + n0 * 1024, 1024, c00, c01, c10, c11);
  int lane = threadIdx.x & 31, half = lane >> 4, n = lane & 15;
  fc2_store(c00, c10, n0 + n,      m0, half, bias, out);
  fc2_store(c01, c11, n0 + 16 + n, m0, half, bias, out);
}

// ---------------------------------------------------------------------------
extern "C" void kernel_launch(void* const* d_in, const int* in_sizes, int n_in,
                              void* d_out, int out_size, void* d_ws, size_t ws_size,
                              hipStream_t stream) {
  (void)in_sizes; (void)n_in; (void)out_size; (void)ws_size;
  const float* img0    = (const float*)d_in[0];
  const float* img1    = (const float*)d_in[1];
  const float* mask    = (const float*)d_in[2];
  const float* w_patch = (const float*)d_in[3];
  const float* b_patch = (const float*)d_in[4];
  const float* w_kv    = (const float*)d_in[5];
  const float* b_kv    = (const float*)d_in[6];
  const float* w_qv    = (const float*)d_in[7];
  const float* b_qv    = (const float*)d_in[8];
  const float* w_fc1   = (const float*)d_in[9];
  const float* b_fc1   = (const float*)d_in[10];
  const float* w_fc2   = (const float*)d_in[11];
  const float* b_fc2   = (const float*)d_in[12];
  float* out = (float*)d_out;

  char* base = (char*)d_ws;
  size_t off = 0;
  auto take = [&](size_t bytes) -> char* {
    char* p = base + off;
    off += (bytes + 255) & ~(size_t)255;
    return p;
  };
  // persistent region
  _Float16* wpt   = (_Float16*)take(256u * 192u * 2u);     // w_patch^T (N-major)
  _Float16* wkvt  = (_Float16*)take(512u * 256u * 2u);
  _Float16* wqvt  = (_Float16*)take(512u * 256u * 2u);
  _Float16* wf1t  = (_Float16*)take(1024u * 256u * 2u);
  _Float16* wf2t  = (_Float16*)take(192u * 1024u * 2u);
  float*    biasb = (float*)   take(8192u * 4u);
  _Float16* vt    = (_Float16*)take(64u * 32u * 1024u * 2u);   // V transposed
  _Float16* aout  = (_Float16*)take(8192u * 256u * 2u);        // attention out
  // reusable region (dead after attention; MLP hidden aliases it)
  _Float16* pat0  = (_Float16*)take(8192u * 192u * 2u);
  _Float16* pat1  = (_Float16*)take(8192u * 192u * 2u);
  _Float16* x0h   = (_Float16*)take(8192u * 256u * 2u);
  _Float16* x1h   = (_Float16*)take(8192u * 256u * 2u);
  _Float16* kbuf  = (_Float16*)take(8192u * 32u * 2u * 8u);    // (b,h,tok,32)
  _Float16* qbuf  = (_Float16*)take(8192u * 32u * 2u * 8u);
  _Float16* hbuf  = pat0;   // 16 MB, overlays pat0..kbuf (all dead by then)

  // weight staging (cast + transpose to N-major)
  cast_transpose_kernel<<<(192 * 256 + 255) / 256, 256, 0, stream>>>(w_patch, wpt, 192, 256);
  cast_transpose_kernel<<<(256 * 512 + 255) / 256, 256, 0, stream>>>(w_kv, wkvt, 256, 512);
  cast_transpose_kernel<<<(256 * 512 + 255) / 256, 256, 0, stream>>>(w_qv, wqvt, 256, 512);
  cast_transpose_kernel<<<(256 * 1024 + 255) / 256, 256, 0, stream>>>(w_fc1, wf1t, 256, 1024);
  cast_transpose_kernel<<<(1024 * 192 + 255) / 256, 256, 0, stream>>>(w_fc2, wf2t, 1024, 192);

  patchify_kernel<<<6144, 256, 0, stream>>>(img0, pat0);
  patchify_kernel<<<6144, 256, 0, stream>>>(img1, pat1);
  maskbias_kernel<<<32, 256, 0, stream>>>(mask, biasb);

  embed_kernel<<<512, 128, 0, stream>>>(pat0, wpt, b_patch, x0h);
  embed_kernel<<<512, 128, 0, stream>>>(pat1, wpt, b_patch, x1h);

  kv_kernel<<<1024, 128, 0, stream>>>(x0h, wkvt, b_kv, kbuf, vt);
  q_kernel<<<512, 128, 0, stream>>>(x1h, wqvt, b_qv, qbuf);

  attn_kernel<<<512, 128, 0, stream>>>(qbuf, kbuf, vt, biasb, aout);

  fc1_kernel<<<2048, 128, 0, stream>>>(aout, wf1t, b_fc1, hbuf);
  fc2_kernel<<<384, 128, 0, stream>>>(hbuf, wf2t, b_fc2, out);
}
